// EnhancedAdaptiveFeature_24919400251995
// MI455X (gfx1250) — matrix operand order repaired and checked by reading, not compiled, not run on hardware
//
#include <hip/hip_runtime.h>
#include <hip/hip_bf16.h>
#include <stdint.h>

// ---------------------------------------------------------------------------
// MI455X (gfx1250) implementation of EnhancedAdaptiveFeature.
//  - Dense GEMMs ( ChebConv k-terms, FE MLP ) use v_wmma_f32_16x16x32_bf16
//    with fp32 accumulation; 256-thread (8-wave wave32) blocks, 128x64 tile.
//  - Sparse Laplacian props use float4 gather + global_atomic_add_f32 scatter.
// ---------------------------------------------------------------------------

typedef __attribute__((ext_vector_type(16))) __bf16 v16bf;
typedef __attribute__((ext_vector_type(8)))  float  v8f;

#define CC   128
#define BEPS 1e-5f

__device__ __forceinline__ unsigned short f2bf(float f) {
  unsigned int u = __float_as_uint(f);
  u += 0x7FFFu + ((u >> 16) & 1u);           // round-to-nearest-even
  return (unsigned short)(u >> 16);
}

union FragU { uint4 q[2]; v16bf v; };

// A-matrix 16-bit 16x32 layout (ISA 7.12.2): lanes 0-15 hold K={0..7,16..23},
// lanes 16-31 hold K={8..15,24..31}; two contiguous 16B chunks per lane.
__device__ __forceinline__ v16bf loadFragA(const unsigned short* rowp, int hi) {
  int k0 = hi ? 8 : 0;
  FragU u;
  u.q[0] = *(const uint4*)(rowp + k0);
  u.q[1] = *(const uint4*)(rowp + k0 + 16);
  return u.v;
}
// B-matrix (per-lane column): lanes 0-15 K={0..7,16..23}, lanes 16-31
// K={8..15,24..31} -> two 16B chunks per lane (matches A).
__device__ __forceinline__ v16bf loadFragB(const unsigned short* rowp, int hi) {
  int k0 = hi ? 8 : 0;
  FragU u;
  u.q[0] = *(const uint4*)(rowp + k0);
  u.q[1] = *(const uint4*)(rowp + k0 + 16);
  return u.v;
}

// ---------------------------------------------------------------------------
// Fused GEMM: Y = act(bn(A@W [+Y if accum] + bias)) [+resid]
// A:[M,K] f32, W:[K,N] f32, Y:[M,N] f32.  K multiple of 32.
// flags: bit0 = accumulate into Y; bits[2:1] = act (0 none, 1 relu, 2 sigmoid)
// ---------------------------------------------------------------------------
__global__ __launch_bounds__(256)
void gemm_wmma_bf16(const float* __restrict__ A, const float* __restrict__ W,
                    const float* __restrict__ bias,
                    const float* __restrict__ bng, const float* __restrict__ bnb,
                    const float* __restrict__ bnm, const float* __restrict__ bnv,
                    const float* __restrict__ resid,
                    int flags, float* __restrict__ Y,
                    int M, int K, int N) {
  __shared__ __align__(16) unsigned short As[128][40];  // 128x32 bf16, +8 pad
  __shared__ __align__(16) unsigned short Bs[64][40];   // 64(N) x 32(K) bf16

  const int t     = threadIdx.x;
  const int wave  = t >> 5;
  const int lane  = t & 31;
  const int waveM = wave & 3;           // 4 waves along M (32 rows each)
  const int waveN = wave >> 2;          // 2 waves along N (32 cols each)
  const int lhalf = lane & 15;
  const int hisel = (lane >= 16);

  const int rowBase = blockIdx.x * 128;
  const int colBase = blockIdx.y * 64;

  v8f acc[2][2] = {};

  for (int kk = 0; kk < K; kk += 32) {
    // ---- stage A tile (128x32) : 1024 float4 loads across 256 threads ----
#pragma unroll
    for (int i = 0; i < 4; ++i) {
      int idx = i * 256 + t;            // 0..1023
      int r   = idx >> 3;               // row in tile
      int q   = idx & 7;                // float4 slot in the 32-wide K chunk
      float4 v = make_float4(0.f, 0.f, 0.f, 0.f);
      int grow = rowBase + r;
      if (grow < M) v = *(const float4*)(A + (size_t)grow * K + kk + q * 4);
      unsigned short* dst = &As[r][q * 4];
      dst[0] = f2bf(v.x); dst[1] = f2bf(v.y);
      dst[2] = f2bf(v.z); dst[3] = f2bf(v.w);
    }
    // ---- stage B tile (32x64) transposed -> Bs[n][k] ----
#pragma unroll
    for (int i = 0; i < 2; ++i) {
      int idx = i * 256 + t;            // 0..511
      int kr  = idx >> 4;               // 0..31  (k within chunk)
      int nq  = idx & 15;               // 16 quads of n
      const float* wrow = W + (size_t)(kk + kr) * N + colBase;
#pragma unroll
      for (int j = 0; j < 4; ++j) {
        int n = nq * 4 + j;
        float v = (colBase + n < N) ? wrow[n] : 0.f;
        Bs[n][kr] = f2bf(v);
      }
    }
    __syncthreads();

#pragma unroll
    for (int sm = 0; sm < 2; ++sm) {
      int m = waveM * 32 + sm * 16 + lhalf;
      v16bf aF = loadFragA(&As[m][0], hisel);
#pragma unroll
      for (int sn = 0; sn < 2; ++sn) {
        int n = waveN * 32 + sn * 16 + lhalf;
        v16bf bF = loadFragB(&Bs[n][0], hisel);
        acc[sm][sn] = __builtin_amdgcn_wmma_f32_16x16x32_bf16(
            false, aF, false, bF, (short)0, acc[sm][sn], false, false);
      }
    }
    __syncthreads();
  }

  // ---- epilogue ----
  const int accum = flags & 1;
  const int act   = (flags >> 1) & 3;
#pragma unroll
  for (int sm = 0; sm < 2; ++sm) {
#pragma unroll
    for (int sn = 0; sn < 2; ++sn) {
#pragma unroll
      for (int r = 0; r < 8; ++r) {
        int mloc = r + (hisel ? 8 : 0);          // C/D layout: vgpr r -> M
        int nloc = lhalf;                        // lane -> N
        int grow = rowBase + waveM * 32 + sm * 16 + mloc;
        int gcol = colBase + waveN * 32 + sn * 16 + nloc;
        if (grow < M && gcol < N) {
          size_t o = (size_t)grow * N + gcol;
          float v = acc[sm][sn][r];
          if (accum) v += Y[o];
          if (bias)  v += bias[gcol];
          if (bng)   v = (v - bnm[gcol]) * rsqrtf(bnv[gcol] + BEPS) * bng[gcol] + bnb[gcol];
          if (act == 1)      v = fmaxf(v, 0.f);
          else if (act == 2) v = 1.f / (1.f + expf(-v));
          if (resid) v += resid[o];
          Y[o] = v;
        }
      }
    }
  }
}

// ---------------------------------------------------------------------------
// Small helper kernels
// ---------------------------------------------------------------------------
__global__ void fill_zero_k(float* p, long long n) {
  long long i = (long long)blockIdx.x * blockDim.x + threadIdx.x;
  if (i < n) p[i] = 0.f;
}

__global__ void colsum_k(const float* __restrict__ x, float* __restrict__ msum, int Nn) {
  int c = threadIdx.x;                 // 128 threads = channels
  float s = 0.f;
  for (int i = blockIdx.x; i < Nn; i += gridDim.x) s += x[(size_t)i * CC + c];
  atomicAdd(&msum[c], s);
}

__global__ void ca_mlp_k(const float* __restrict__ msum,
                         const float* __restrict__ W1, const float* __restrict__ b1,
                         const float* __restrict__ g,  const float* __restrict__ b,
                         const float* __restrict__ mu, const float* __restrict__ var,
                         const float* __restrict__ W2, const float* __restrict__ b2,
                         float* __restrict__ cw, int Nn) {
  __shared__ float m[128];
  __shared__ float h[64];
  int t = threadIdx.x;
  m[t] = msum[t] / (float)Nn;
  __syncthreads();
  if (t < 64) {
    float s = b1[t];
    for (int c = 0; c < 128; ++c) s += m[c] * W1[c * 64 + t];
    s = (s - mu[t]) * rsqrtf(var[t] + BEPS) * g[t] + b[t];
    h[t] = fmaxf(s, 0.f);
  }
  __syncthreads();
  float s = b2[t];
  for (int j = 0; j < 64; ++j) s += h[j] * W2[j * 128 + t];
  cw[t] = 1.f / (1.f + expf(-s));
}

__global__ void degree_k(const int* __restrict__ row, float* __restrict__ deg, int E) {
  int e = blockIdx.x * blockDim.x + threadIdx.x;
  if (e < E) atomicAdd(&deg[row[e]], 1.f);
}

__global__ void dinv_k(const float* __restrict__ deg, float* __restrict__ dinv, int Nn) {
  int i = blockIdx.x * blockDim.x + threadIdx.x;
  if (i < Nn) {
    float d = deg[i];
    dinv[i] = (d > 0.f) ? rsqrtf(fmaxf(d, 1.f)) : 0.f;
  }
}

__global__ void edgew_k(const int* __restrict__ row, const int* __restrict__ col,
                        const float* __restrict__ dinv, float* __restrict__ w, int E) {
  int e = blockIdx.x * blockDim.x + threadIdx.x;
  if (e < E) w[e] = -dinv[row[e]] * dinv[col[e]];
}

__global__ void chanscale_k(const float* __restrict__ x, const float* __restrict__ cw,
                            float* __restrict__ y, long long total) {
  long long i = (long long)blockIdx.x * blockDim.x + threadIdx.x;
  if (i < total) y[i] = x[i] * cw[i & (CC - 1)];
}

__global__ void rowscale_k(float* __restrict__ y, const float* __restrict__ sw, long long total) {
  long long i = (long long)blockIdx.x * blockDim.x + threadIdx.x;
  if (i < total) y[i] *= sw[i >> 7];   // C = 128
}

// out[col[e]][:] += w[e] * h[row[e]][:]   (one float4 per thread)
__global__ void prop_scatter_k(const float* __restrict__ h, const int* __restrict__ row,
                               const int* __restrict__ col, const float* __restrict__ w,
                               float* __restrict__ out, long long total, int C, int qshift) {
  long long gid = (long long)blockIdx.x * blockDim.x + threadIdx.x;
  if (gid >= total) return;
  int e = (int)(gid >> qshift);
  int q = (int)(gid & ((1 << qshift) - 1));
  int r = row[e], c = col[e];
  float we = w[e];
  float4 hv = *(const float4*)(h + (size_t)r * C + q * 4);
  float* op = out + (size_t)c * C + q * 4;
  atomicAdd(op + 0, we * hv.x);
  atomicAdd(op + 1, we * hv.y);
  atomicAdd(op + 2, we * hv.z);
  atomicAdd(op + 3, we * hv.w);
}

// y = 2*y - x0  (Chebyshev recurrence combine, in place on prop output)
__global__ void cheb_comb_k(float* __restrict__ y, const float* __restrict__ x0, long long n) {
  long long i = (long long)blockIdx.x * blockDim.x + threadIdx.x;
  if (i < n) y[i] = 2.f * y[i] - x0[i];
}

// ---------------------------------------------------------------------------
extern "C" void kernel_launch(void* const* d_in, const int* in_sizes, int n_in,
                              void* d_out, int out_size, void* d_ws, size_t ws_size,
                              hipStream_t stream) {
  // Inputs flattened in setup_inputs() dict order (recursive insertion order).
  const float* x    = (const float*)d_in[0];
  const float* ca1W = (const float*)d_in[1];
  const float* ca1b = (const float*)d_in[2];
  const float* cag  = (const float*)d_in[3];
  const float* cabt = (const float*)d_in[4];
  const float* cam  = (const float*)d_in[5];
  const float* cav  = (const float*)d_in[6];
  const float* ca2W = (const float*)d_in[7];
  const float* ca2b = (const float*)d_in[8];
  const float* sp1W = (const float*)d_in[9];    // [4,128,256]
  const float* sp1b = (const float*)d_in[10];
  const float* spg  = (const float*)d_in[11];
  const float* spbt = (const float*)d_in[12];
  const float* spm  = (const float*)d_in[13];
  const float* spv  = (const float*)d_in[14];
  const float* sp2W = (const float*)d_in[15];   // [3,256,1]
  const float* sp2b = (const float*)d_in[16];
  const float* fe1W = (const float*)d_in[17];   // [128,512]
  const float* fe1b = (const float*)d_in[18];
  const float* f1g  = (const float*)d_in[19];
  const float* f1b  = (const float*)d_in[20];
  const float* f1m  = (const float*)d_in[21];
  const float* f1v  = (const float*)d_in[22];
  const float* fe2W = (const float*)d_in[23];   // [512,256]
  const float* fe2b = (const float*)d_in[24];
  const float* f2g  = (const float*)d_in[25];
  const float* f2b  = (const float*)d_in[26];
  const float* f2m  = (const float*)d_in[27];
  const float* f2v  = (const float*)d_in[28];
  const float* fe3W = (const float*)d_in[29];   // [256,128]
  const float* fe3b = (const float*)d_in[30];
  const int*   eidx = (const int*)d_in[31];

  const int Nn = in_sizes[0] / CC;              // 50000
  const int E  = in_sizes[31] / 2;              // 600000
  const int* row = eidx;
  const int* col = eidx + E;

  // ---- workspace layout (f32, 256-element aligned) ----
  float* ws = (float*)d_ws;
  size_t off = 0;
  auto alloc = [&](size_t n) { size_t o = off; off += (n + 255) & ~(size_t)255; return ws + o; };
  float* msum = alloc(CC);
  float* cw   = alloc(CC);
  float* deg  = alloc(Nn);
  float* dinv = alloc(Nn);
  float* wE   = alloc(E);
  float* x1   = alloc((size_t)Nn * 128);        // Tx0 for conv1, later x2
  float* T1   = alloc((size_t)Nn * 128);
  float* T2   = alloc((size_t)Nn * 128);
  float* T3   = alloc((size_t)Nn * 128);
  float* s    = alloc((size_t)Nn * 256);        // conv1 output / conv2 Tx0
  float* U1   = alloc((size_t)Nn * 256);
  float* U2   = alloc((size_t)Nn * 256);        // later reused as fe2 output
  float* y2   = alloc(Nn);                      // sw
  float* h1   = alloc((size_t)Nn * 512);

  auto fill = [&](float* p, long long n) {
    fill_zero_k<<<dim3((unsigned)((n + 255) / 256)), 256, 0, stream>>>(p, n);
  };
  auto gemm = [&](const float* A, const float* W, const float* bias,
                  const float* g, const float* b, const float* mu, const float* var,
                  const float* resid, int flags, float* Y, int M, int K, int N) {
    dim3 grid((M + 127) / 128, (N + 63) / 64);
    gemm_wmma_bf16<<<grid, 256, 0, stream>>>(A, W, bias, g, b, mu, var, resid, flags, Y, M, K, N);
  };
  auto prop = [&](const float* h, float* out, int C, int qshift) {
    long long total = (long long)E << qshift;
    prop_scatter_k<<<dim3((unsigned)((total + 255) / 256)), 256, 0, stream>>>(
        h, row, col, wE, out, total, C, qshift);
  };

  // ---- 1. channel attention ----
  fill(msum, CC);
  colsum_k<<<512, 128, 0, stream>>>(x, msum, Nn);
  ca_mlp_k<<<1, 128, 0, stream>>>(msum, ca1W, ca1b, cag, cabt, cam, cav, ca2W, ca2b, cw, Nn);
  long long totNC = (long long)Nn * CC;
  chanscale_k<<<dim3((unsigned)((totNC + 255) / 256)), 256, 0, stream>>>(x, cw, x1, totNC);

  // ---- 2. scaled-Laplacian edge weights ----
  fill(deg, Nn);
  degree_k<<<(E + 255) / 256, 256, 0, stream>>>(row, deg, E);
  dinv_k<<<(Nn + 255) / 256, 256, 0, stream>>>(deg, dinv, Nn);
  edgew_k<<<(E + 255) / 256, 256, 0, stream>>>(row, col, dinv, wE, E);

  // ---- 3. ChebConv1 (128 -> 256, K=4) -> BN -> ReLU  into s ----
  gemm(x1, sp1W + 0 * 128 * 256, nullptr, nullptr, nullptr, nullptr, nullptr,
       nullptr, /*flags*/0, s, Nn, 128, 256);
  fill(T1, totNC);
  prop(x1, T1, 128, 5);                                   // Tx1 = L_hat @ x1
  gemm(T1, sp1W + 1 * 128 * 256, nullptr, nullptr, nullptr, nullptr, nullptr,
       nullptr, 1, s, Nn, 128, 256);
  fill(T2, totNC);
  prop(T1, T2, 128, 5);
  cheb_comb_k<<<dim3((unsigned)((totNC + 255) / 256)), 256, 0, stream>>>(T2, x1, totNC);
  gemm(T2, sp1W + 2 * 128 * 256, nullptr, nullptr, nullptr, nullptr, nullptr,
       nullptr, 1, s, Nn, 128, 256);
  fill(T3, totNC);
  prop(T2, T3, 128, 5);
  cheb_comb_k<<<dim3((unsigned)((totNC + 255) / 256)), 256, 0, stream>>>(T3, T1, totNC);
  gemm(T3, sp1W + 3 * 128 * 256, sp1b, spg, spbt, spm, spv,
       nullptr, 1 | (1 << 1), s, Nn, 128, 256);           // accum + bias + BN + relu

  // ---- 4. ChebConv2 (256 -> 1, K=3) -> sigmoid  into y2 (sw) ----
  long long totN2 = (long long)Nn * 256;
  gemm(s, sp2W + 0 * 256, nullptr, nullptr, nullptr, nullptr, nullptr,
       nullptr, 0, y2, Nn, 256, 1);
  fill(U1, totN2);
  prop(s, U1, 256, 6);
  gemm(U1, sp2W + 1 * 256, nullptr, nullptr, nullptr, nullptr, nullptr,
       nullptr, 1, y2, Nn, 256, 1);
  fill(U2, totN2);
  prop(U1, U2, 256, 6);
  cheb_comb_k<<<dim3((unsigned)((totN2 + 255) / 256)), 256, 0, stream>>>(U2, s, totN2);
  gemm(U2, sp2W + 2 * 256, sp2b, nullptr, nullptr, nullptr, nullptr,
       nullptr, 1 | (2 << 1), y2, Nn, 256, 1);            // accum + bias + sigmoid

  // ---- 5. x2 = x1 * sw (in place) ----
  rowscale_k<<<dim3((unsigned)((totNC + 255) / 256)), 256, 0, stream>>>(x1, y2, totNC);

  // ---- 6. feature-enhancement MLP + residual ----
  gemm(x1, fe1W, fe1b, f1g, f1b, f1m, f1v, nullptr, (1 << 1), h1, Nn, 128, 512);
  gemm(h1, fe2W, fe2b, f2g, f2b, f2m, f2v, nullptr, (1 << 1), U2, Nn, 512, 256);
  gemm(U2, fe3W, fe3b, nullptr, nullptr, nullptr, nullptr,
       x1 /*resid = x2*/, 0, (float*)d_out, Nn, 256, 128);
}
